// Head_46703474377397
// MI455X (gfx1250) — matrix operand order repaired and checked by reading, not compiled, not run on hardware
//
#include <hip/hip_runtime.h>
#include <hip/hip_bf16.h>

// ---------------------------------------------------------------------------
// Single-head attention (no causal mask), B=8 T=2048 E=1024 H=64, f32 in/out.
// bf16 WMMA (v_wmma_f32_16x16x32_bf16) everywhere; flash-attention style
// fused softmax computing S^T / O^T so softmax is lane-local (lane = query).
// ---------------------------------------------------------------------------

typedef __attribute__((ext_vector_type(16))) __bf16 v16bf;
typedef __attribute__((ext_vector_type(8)))  __bf16 v8bf;
typedef __attribute__((ext_vector_type(8)))  float  v8f;
typedef __attribute__((ext_vector_type(4)))  int    v4i;

#define BB 8
#define TT 2048
#define NE 1024
#define HH 64
// softmax evaluated in exp2 domain: s * (H^-0.5 * log2(e))
#define C2EXP 0.18033688011112042f

// ---- operand loaders ------------------------------------------------------
// A operand (16x32 bf16, M in lanes): lane<16 holds K 0..15, lane>=16
// holds K 16..31 (contiguous). Caller passes rowBase + koffA (0 or 16).
__device__ __forceinline__ v16bf load_a(const __bf16* p) {
  v8bf lo = *(const v8bf*)p;
  v8bf hi = *(const v8bf*)(p + 8);
  v16bf r;
#pragma unroll
  for (int i = 0; i < 8; ++i) { r[i] = lo[i]; r[i + 8] = hi[i]; }
  return r;
}
// B operand (32x16 bf16, N in lanes): lane<16 holds K 0..15, lane>=16 holds
// K 16..31 (contiguous). Caller passes colBase + koffB (0 or 16).
__device__ __forceinline__ v16bf load_b(const __bf16* p) {
  return *(const v16bf*)p;   // 32B contiguous, 32B aligned by construction
}
__device__ __forceinline__ v16bf load_b_f32(const float* p) {
  v16bf r;
#pragma unroll
  for (int i = 0; i < 16; ++i) r[i] = (__bf16)p[i];
  return r;
}
__device__ __forceinline__ v8f vzero8() {
  v8f z = {0.f, 0.f, 0.f, 0.f, 0.f, 0.f, 0.f, 0.f};
  return z;
}
__device__ __forceinline__ v8f wmma_bf16(v16bf a, v16bf b, v8f c) {
  return __builtin_amdgcn_wmma_f32_16x16x32_bf16(false, a, false, b,
                                                 (short)0, c, false, false);
}

// ---- kernel 0: W[1024,64] f32 -> Wt[64,1024] bf16, all three matrices -----
__global__ __launch_bounds__(256) void wtrans_kernel(
    const float* __restrict__ Wq, const float* __restrict__ Wk,
    const float* __restrict__ Wv, __bf16* __restrict__ Wt) {
  const int idx = blockIdx.x * 256 + threadIdx.x;     // < 3*64*1024
  const int k = idx & (NE - 1);
  const int n = (idx >> 10) & (HH - 1);
  const int m = idx >> 16;
  const float* W = (m == 0) ? Wq : ((m == 1) ? Wk : Wv);
  Wt[idx] = (__bf16)W[(size_t)k * HH + n];
}

// ---- kernel 1: QKV projection --------------------------------------------
// Out^T[h,t] tiles = Wt(A) x x^T(B); one wave = 16 rows of x, all 64 output
// cols for Q, K and V (x read exactly once from HBM). Stores Q,K row-major
// bf16 [B*T,64]; V transposed bf16 [B,64,T].
__global__ __launch_bounds__(256) void proj_kernel(
    const float* __restrict__ x, const __bf16* __restrict__ Wt,
    __bf16* __restrict__ Qbf, __bf16* __restrict__ Kbf,
    __bf16* __restrict__ Vt) {
  const int lane  = threadIdx.x & 31;
  const int wave  = threadIdx.x >> 5;
  const int tile  = blockIdx.x * 8 + wave;            // 0..1023
  const int t0    = tile * 16;                        // row in [0, B*T)
  const int lr    = lane & 15;
  const int koffA = (lane < 16) ? 0 : 16;
  const int koffB = (lane < 16) ? 0 : 16;

  v8f acc[3][4];
#pragma unroll
  for (int m = 0; m < 3; ++m)
#pragma unroll
    for (int h = 0; h < 4; ++h) acc[m][h] = vzero8();

  const float* xrow = x + (size_t)(t0 + lr) * NE;

#pragma unroll 1
  for (int k0 = 0; k0 < NE; k0 += 32) {
    v16bf xb = load_b_f32(xrow + k0 + koffB);         // shared B operand
#pragma unroll
    for (int m = 0; m < 3; ++m) {
      const __bf16* wb = Wt + (size_t)m * (HH * NE);
#pragma unroll
      for (int h = 0; h < 4; ++h) {
        v16bf wa = load_a(wb + (size_t)(h * 16 + lr) * NE + k0 + koffA);
        acc[m][h] = wmma_bf16(wa, xb, acc[m][h]);
      }
    }
  }

  // D layout: lane = t column (t0+lr); rows M -> h = h*16 + hoff + i
  const int t    = t0 + lr;
  const int b    = t / TT;
  const int tq   = t & (TT - 1);
  const int hoff = (lane < 16) ? 0 : 8;

  __bf16* qrow = Qbf + (size_t)t * HH + hoff;
  __bf16* krow = Kbf + (size_t)t * HH + hoff;
#pragma unroll
  for (int h = 0; h < 4; ++h) {
    v8bf qv, kv;
#pragma unroll
    for (int i = 0; i < 8; ++i) {
      qv[i] = (__bf16)acc[0][h][i];
      kv[i] = (__bf16)acc[1][h][i];
    }
    *(v8bf*)(qrow + h * 16) = qv;
    *(v8bf*)(krow + h * 16) = kv;
  }
  __bf16* vb = Vt + (size_t)b * HH * TT + tq;
#pragma unroll
  for (int h = 0; h < 4; ++h)
#pragma unroll
    for (int i = 0; i < 8; ++i)
      vb[(size_t)(h * 16 + hoff + i) * TT] = (__bf16)acc[2][h][i];
}

// ---- kernel 2: fused attention (online softmax, 64-key tiles) -------------
// Per wave: 16 queries. Loop over 64-key tiles:
//   S^T(64x16) = K(A) x Q^T(B)   -> 4 sub-tiles; lane holds one query column
//   exp2-domain lane-local max/sum (+1 shfl_xor16 each), alpha-rescale O^T
//   P^T built with 8 packed-u32 cross-half shuffles per 32 keys
//   O^T(64x16) += Vt(A) x P^T(B)
__global__ __launch_bounds__(256) void attn_kernel(
    const __bf16* __restrict__ Qbf, const __bf16* __restrict__ Kbf,
    const __bf16* __restrict__ Vt, float* __restrict__ out) {
  const int lane  = threadIdx.x & 31;
  const int wave  = threadIdx.x >> 5;
  const int tile  = blockIdx.x * 8 + wave;            // 0..1023
  const int b     = tile >> 7;                        // / (T/16 = 128)
  const int q0    = (tile & 127) * 16;
  const int lr    = lane & 15;
  const int koffA = (lane < 16) ? 0 : 16;
  const int koffB = (lane < 16) ? 0 : 16;
  const bool lo   = (lane < 16);
  const float NEG_INF = -__builtin_inff();

  const __bf16* qrow = Qbf + ((size_t)b * TT + q0 + lr) * HH;
  v16bf qb0 = load_b(qrow + 0  + koffB);              // h-chunk 0..31
  v16bf qb1 = load_b(qrow + 32 + koffB);              // h-chunk 32..63

  v8f acc[4];
#pragma unroll
  for (int h = 0; h < 4; ++h) acc[h] = vzero8();
  float row_max = NEG_INF;                            // exp2 domain
  float row_sum = 0.f;

  const __bf16* Kb = Kbf + (size_t)b * TT * HH;
  const __bf16* Vb = Vt + (size_t)b * HH * TT;

#pragma unroll 1
  for (int k0 = 0; k0 < TT; k0 += 64) {
    if (k0 + 64 < TT) {                               // global_prefetch_b8
      __builtin_prefetch(Kb + (size_t)(k0 + 64 + lr) * HH, 0, 1);
      __builtin_prefetch(Vb + (size_t)lr * TT + k0 + 64, 0, 1);
    }
    // ---- S^T sub-tiles: st[m] covers keys k0+16m .. k0+16m+15 ----
    v8f st[4];
#pragma unroll
    for (int m = 0; m < 4; ++m) {
      st[m] = vzero8();
      v16bf ka = load_a(Kb + (size_t)(k0 + 16 * m + lr) * HH + 0 + koffA);
      st[m] = wmma_bf16(ka, qb0, st[m]);
      ka = load_a(Kb + (size_t)(k0 + 16 * m + lr) * HH + 32 + koffA);
      st[m] = wmma_bf16(ka, qb1, st[m]);
    }
    // ---- online softmax (exp2 domain); each lane owns one query ----
    float p[4][8];
    float vmax = NEG_INF;
#pragma unroll
    for (int m = 0; m < 4; ++m)
#pragma unroll
      for (int i = 0; i < 8; ++i) {
        p[m][i] = st[m][i] * C2EXP;
        vmax = fmaxf(vmax, p[m][i]);
      }
    vmax = fmaxf(vmax, __shfl_xor(vmax, 16, 32));
    const float mnew  = fmaxf(row_max, vmax);
    const float alpha = __builtin_amdgcn_exp2f(row_max - mnew);
    float lsum = 0.f;
#pragma unroll
    for (int m = 0; m < 4; ++m)
#pragma unroll
      for (int i = 0; i < 8; ++i) {
        p[m][i] = __builtin_amdgcn_exp2f(p[m][i] - mnew);
        lsum += p[m][i];
      }
    lsum += __shfl_xor(lsum, 16, 32);
    row_sum = row_sum * alpha + lsum;
    row_max = mnew;
#pragma unroll
    for (int h = 0; h < 4; ++h)
#pragma unroll
      for (int i = 0; i < 8; ++i) acc[h][i] *= alpha;

    // ---- P^T B operands: pb[c] covers keys k0+32c .. k0+32c+31 ----
    // Convert to bf16 pairs first, exchange packed u32 across lane halves.
    // lane<16 : elements[0..7]=own st[2c], [8..15]=partner st[2c]
    // lane>=16: elements[0..7]=partner st[2c+1], [8..15]=own st[2c+1]
    v16bf pb[2];
#pragma unroll
    for (int c = 0; c < 2; ++c) {
      v8bf e8, o8;
#pragma unroll
      for (int i = 0; i < 8; ++i) {
        e8[i] = (__bf16)p[2 * c][i];
        o8[i] = (__bf16)p[2 * c + 1][i];
      }
      v4i pe = __builtin_bit_cast(v4i, e8);
      v4i po = __builtin_bit_cast(v4i, o8);
      v4i lohalf, hihalf;
#pragma unroll
      for (int j = 0; j < 4; ++j) {
        const int xe = __shfl_xor(pe[j], 16, 32);
        const int xo = __shfl_xor(po[j], 16, 32);
        lohalf[j] = lo ? pe[j] : xo;
        hihalf[j] = lo ? xe : po[j];
      }
      v8bf l8 = __builtin_bit_cast(v8bf, lohalf);
      v8bf h8 = __builtin_bit_cast(v8bf, hihalf);
#pragma unroll
      for (int i = 0; i < 8; ++i) { pb[c][i] = l8[i]; pb[c][i + 8] = h8[i]; }
    }
    // ---- O^T += Vt x P^T (inner dim 64 = 2 chunks of 32) ----
#pragma unroll
    for (int h = 0; h < 4; ++h) {
      v16bf va = load_a(Vb + (size_t)(h * 16 + lr) * TT + k0 + koffA);
      acc[h] = wmma_bf16(va, pb[0], acc[h]);
      va = load_a(Vb + (size_t)(h * 16 + lr) * TT + k0 + 32 + koffA);
      acc[h] = wmma_bf16(va, pb[1], acc[h]);
    }
  }

  // D layout: lane = q column; rows -> h = h*16 + (lane<16?0:8) + i (contig)
  const float inv = 1.0f / row_sum;
  float* orow = out + ((size_t)b * TT + q0 + lr) * HH + ((lane < 16) ? 0 : 8);
#pragma unroll
  for (int h = 0; h < 4; ++h) {
    v8f r;
#pragma unroll
    for (int i = 0; i < 8; ++i) r[i] = acc[h][i] * inv;
    *(v8f*)(orow + h * 16) = r;
  }
}

// ---------------------------------------------------------------------------
extern "C" void kernel_launch(void* const* d_in, const int* in_sizes, int n_in,
                              void* d_out, int out_size, void* d_ws,
                              size_t ws_size, hipStream_t stream) {
  const float* x  = (const float*)d_in[0];
  const float* Wq = (const float*)d_in[1];
  const float* Wk = (const float*)d_in[2];
  const float* Wv = (const float*)d_in[3];
  float* out = (float*)d_out;

  // workspace (bf16): Wt[3*64*1024] | Q[B*T*64] | K[B*T*64] | Vt[B*64*T]
  __bf16* ws  = (__bf16*)d_ws;
  __bf16* Wt  = ws;
  __bf16* Qbf = Wt + (size_t)3 * HH * NE;
  __bf16* Kbf = Qbf + (size_t)BB * TT * HH;
  __bf16* Vt  = Kbf + (size_t)BB * TT * HH;

  wtrans_kernel<<<(3 * HH * NE) / 256, 256, 0, stream>>>(Wq, Wk, Wv, Wt);
  proj_kernel<<<(BB * TT / 16) / 8, 256, 0, stream>>>(x, Wt, Qbf, Kbf, Vt);
  attn_kernel<<<(BB * TT / 16) / 8, 256, 0, stream>>>(Qbf, Kbf, Vt, out);
}
